// Attention_17703855194556
// MI455X (gfx1250) — matrix operand order repaired and checked by reading, compiled, not run on hardware
//
#include <hip/hip_runtime.h>
#include <hip/hip_bf16.h>

// Problem constants (from reference): B=2, S=2048, D=1024, H=16, HD=64
#define S_LEN 2048
#define DIM   1024
#define NH    16
#define HD_   64
#define NB    2
#define SCALE_F 8.0f

typedef __bf16 bf16;
typedef __attribute__((ext_vector_type(16))) __bf16 v16bf;
typedef __attribute__((ext_vector_type(8)))  __bf16 v8bf;
typedef __attribute__((ext_vector_type(8)))  float  v8f;
typedef __attribute__((ext_vector_type(4)))  unsigned int u32x4;
typedef __attribute__((ext_vector_type(8)))  int i32x8;
typedef __attribute__((ext_vector_type(4)))  int i32x4;

#if __has_include(<hip/amd_detail/amd_gfx1250_TDM.h>)
#define TDM_SIX_ARGS 1
#endif

union ABf { v16bf v; v8bf h[2]; };

// ---------------------------------------------------------------------------
// Fragment loaders matching CDNA5 WMMA VGPR layouts (ISA 7.12.2, wave32):
//  A (16x32 bf16, row-major src, stride ld):
//    lane L holds row M = L&15; VGPR0-3 = K[(L>>4)*8 .. +8), VGPR4-7 = +16.
//  B (32x16 bf16) from a transposed matrix BT (row-major [N,K], stride ld):
//    lane L holds col N = L&15; VGPR0-3 = K[(L>>4)*8 .. +8), VGPR4-7 = +16.
//  C/D (16x16 f32): lane L col N = L&15; VGPR r = row r + (L>>4)*8.
// ---------------------------------------------------------------------------
__device__ __forceinline__ v16bf load_a_frag(const bf16* base, int ld, int m0,
                                             int k0, int lane) {
  int m  = m0 + (lane & 15);
  int kk = k0 + ((lane >> 4) << 3);
  const bf16* p = base + (long long)m * ld + kk;
  ABf a;
  a.h[0] = *reinterpret_cast<const v8bf*>(p);
  a.h[1] = *reinterpret_cast<const v8bf*>(p + 16);
  return a.v;
}

__device__ __forceinline__ v16bf load_b_frag(const bf16* bt, int ld, int n0,
                                             int k0, int lane) {
  int n  = n0 + (lane & 15);
  int kk = k0 + ((lane >> 4) << 3);
  const bf16* p = bt + (long long)n * ld + kk;
  ABf b;
  b.h[0] = *reinterpret_cast<const v8bf*>(p);
  b.h[1] = *reinterpret_cast<const v8bf*>(p + 16);
  return b.v;
}

__device__ __forceinline__ v8f wmma_bf16(v16bf a, v16bf b, v8f c) {
  return __builtin_amdgcn_wmma_f32_16x16x32_bf16(false, a, false, b, (short)0,
                                                 c, false, false);
}

// ---------------------------------------------------------------------------
// TDM: async-DMA a 2D bf16 tile [tile_rows x 32] from global into LDS.
// D# packing per ISA 8.3/8.4: group0 = {count/type/addr}, group1 = dims.
//   data_size=1 (2 bytes), tile_dim0=32 elems, tile_dim1=tile_rows,
//   tensor_dim0=row_len, tensor_dim1=num_rows, dim0_stride=row_len.
// Tracked with TENSORcnt; waited via s_wait_tensorcnt.
// ---------------------------------------------------------------------------
__device__ __forceinline__ void tdm_load_2d(unsigned lds_off, const bf16* gaddr,
                                            int tile_rows, int row_len,
                                            int num_rows) {
  unsigned long long ga = (unsigned long long)(uintptr_t)gaddr;
  u32x4 g0 = {0u, 0u, 0u, 0u};
  g0[0] = 1u;                                   // count=1 (valid), user mode
  g0[1] = lds_off;                              // lds_addr (bytes)
  g0[2] = (unsigned)ga;                         // global_addr[31:0]
  g0[3] = (unsigned)(ga >> 32) | 0x80000000u;   // global_addr[56:32] | type=2
  i32x8 g1 = {0, 0, 0, 0, 0, 0, 0, 0};
  g1[0] = 1 << 16;                              // data_size = 2 bytes
  g1[1] = (int)((unsigned)row_len << 16);       // tensor_dim0[15:0]
  g1[2] = (int)(((unsigned)row_len >> 16) |     // tensor_dim0[31:16]
                ((unsigned)num_rows << 16));    // tensor_dim1[15:0]
  g1[3] = (int)(((unsigned)num_rows >> 16) |    // tensor_dim1[31:16]
                (32u << 16));                   // tile_dim0 = 32
  g1[4] = tile_rows;                            // tile_dim1 (tile_dim2=0)
  g1[5] = row_len;                              // tensor_dim0_stride[31:0]
  i32x4 z4 = {0, 0, 0, 0};
#ifdef TDM_SIX_ARGS
  i32x8 z8 = {0, 0, 0, 0, 0, 0, 0, 0};
  __builtin_amdgcn_tensor_load_to_lds(g0, g1, z4, z4, z8, 0);
#else
  __builtin_amdgcn_tensor_load_to_lds(g0, g1, z4, z4, 0);
#endif
}

// ---------------------------------------------------------------------------
// Conversion / transpose kernels (fp32 -> bf16)
// ---------------------------------------------------------------------------
__global__ void convert_x_kernel(const float* __restrict__ x,
                                 bf16* __restrict__ xb) {
  long long i = (long long)blockIdx.x * blockDim.x + threadIdx.x; // 4,194,304
  xb[i] = (bf16)x[i];
}

__global__ void transpose_wattn_kernel(const float* __restrict__ w,
                                       bf16* __restrict__ wt) {
  long long i = (long long)blockIdx.x * blockDim.x + threadIdx.x; // 3,145,728
  int n = (int)(i >> 10);   // 0..3071  (output col of w_attn)
  int k = (int)(i & 1023);  // 0..1023  (input dim)
  wt[i] = (bf16)w[(long long)k * (3 * DIM) + n];
}

__global__ void transpose_wproj_kernel(const float* __restrict__ w,
                                       bf16* __restrict__ wt) {
  long long i = (long long)blockIdx.x * blockDim.x + threadIdx.x; // 1,048,576
  int n = (int)(i >> 10);
  int k = (int)(i & 1023);
  wt[i] = (bf16)w[(long long)k * DIM + n];
}

// ---------------------------------------------------------------------------
// GEMM core: block computes a 64Mx64N tile; K streamed in 32-wide steps.
// A and B tiles (64x32 bf16 = 4 KB each) are DMA'd into double-buffered LDS
// by the Tensor Data Mover (wave 0 issues; EXEC is ignored by TDM), with
// s_wait_tensorcnt(2) so the next tiles' DMA overlaps this step's 16 WMMAs.
// ---------------------------------------------------------------------------
__device__ __forceinline__ void gemm_core(const bf16* __restrict__ Ablk,
                                          const bf16* __restrict__ Bblk,
                                          int a_rows, int b_rows,
                                          bf16* ldsA, bf16* ldsB,
                                          int wave, int lane, v8f acc[4]) {
  unsigned offA = (unsigned)(uintptr_t)ldsA;     // generic->LDS byte offset
  unsigned offB = (unsigned)(uintptr_t)ldsB;
  const unsigned tbytes = 64 * 32 * 2;           // one 4 KB tile
  if (wave == 0) {
    tdm_load_2d(offA, Ablk, 64, DIM, a_rows);
    tdm_load_2d(offB, Bblk, 64, DIM, b_rows);
  }
  int m0l = wave << 4;
#pragma unroll 1
  for (int k0 = 0; k0 < DIM; k0 += 32) {
    int buf = (k0 >> 5) & 1;
    bool more = (k0 + 32) < DIM;
    if (wave == 0) {
      if (more) {
        tdm_load_2d(offA + (buf ^ 1) * tbytes, Ablk + k0 + 32, 64, DIM, a_rows);
        tdm_load_2d(offB + (buf ^ 1) * tbytes, Bblk + k0 + 32, 64, DIM, b_rows);
        __builtin_amdgcn_s_wait_tensorcnt(2);    // current buffers landed
      } else {
        __builtin_amdgcn_s_wait_tensorcnt(0);
      }
    }
    __syncthreads();
    const bf16* At = ldsA + buf * (64 * 32);
    const bf16* Bt = ldsB + buf * (64 * 32);
    v16bf a = load_a_frag(At, 32, m0l, 0, lane);
#pragma unroll
    for (int t = 0; t < 4; ++t) {
      v16bf b = load_b_frag(Bt, 32, t << 4, 0, lane);
      acc[t] = wmma_bf16(a, b, acc[t]);
    }
    __syncthreads();
  }
}

// GEMM 1: qkv[4096,3072] = xb[4096,1024] @ w_attnT + b_attn
// Epilogue scatters into qkvf fp32 laid out as [3][B][H][S][64].
__global__ void gemm_qkv_kernel(const bf16* __restrict__ A,
                                const bf16* __restrict__ BT,
                                const float* __restrict__ bias,
                                float* __restrict__ qkvf) {
  __shared__ bf16 ldsA[2 * 64 * 32];
  __shared__ bf16 ldsB[2 * 64 * 32];
  int lane = threadIdx.x & 31;
  int wave = threadIdx.x >> 5;
  int mblk = blockIdx.x << 6;
  int n0 = blockIdx.y << 6;
  v8f acc[4] = {};
  gemm_core(A + (long long)mblk * DIM, BT + (long long)n0 * DIM, 4096, 3072,
            ldsA, ldsB, wave, lane, acc);

  int m0 = mblk + (wave << 4);
  int hf = lane >> 4, ln = lane & 15;
#pragma unroll
  for (int t = 0; t < 4; ++t) {
    int col = n0 + (t << 4) + ln;          // 0..3071
    float bv = bias[col];
    int which = col >> 10;                 // 0=q,1=k,2=v
    int dcol = col & 1023;
    int h = dcol >> 6, hd = dcol & 63;
#pragma unroll
    for (int r = 0; r < 8; ++r) {
      int row = m0 + r + (hf << 3);        // 0..4095
      int b_ = row >> 11, s = row & 2047;
      long long idx =
          (((long long)((which * NB + b_) * NH + h) * S_LEN + s) << 6) + hd;
      qkvf[idx] = acc[t][r] + bv;
    }
  }
}

// GEMM 2: out[4096,1024] = ob[4096,1024] @ w_projT + b_proj   (fp32 out)
__global__ void gemm_proj_kernel(const bf16* __restrict__ A,
                                 const bf16* __restrict__ BT,
                                 const float* __restrict__ bias,
                                 float* __restrict__ C) {
  __shared__ bf16 ldsA[2 * 64 * 32];
  __shared__ bf16 ldsB[2 * 64 * 32];
  int lane = threadIdx.x & 31;
  int wave = threadIdx.x >> 5;
  int mblk = blockIdx.x << 6;
  int n0 = blockIdx.y << 6;
  v8f acc[4] = {};
  gemm_core(A + (long long)mblk * DIM, BT + (long long)n0 * DIM, 4096, 1024,
            ldsA, ldsB, wave, lane, acc);

  int m0 = mblk + (wave << 4);
  int hf = lane >> 4, ln = lane & 15;
#pragma unroll
  for (int t = 0; t < 4; ++t) {
    int col = n0 + (t << 4) + ln;
    float bv = bias[col];
#pragma unroll
    for (int r = 0; r < 8; ++r) {
      int row = m0 + r + (hf << 3);
      C[(long long)row * DIM + col] = acc[t][r] + bv;
    }
  }
}

// ---------------------------------------------------------------------------
// L2-normalize q and k rows (64 elems each) and convert to bf16.
// One wave per row; 2 elements per lane; full-wave shuffle reduction.
// ---------------------------------------------------------------------------
__global__ void norm_qk_kernel(const float* __restrict__ qkvf,
                               bf16* __restrict__ qb, bf16* __restrict__ kb) {
  int lane = threadIdx.x & 31;
  long long row = (long long)blockIdx.x * (blockDim.x >> 5) + (threadIdx.x >> 5);
  int which = (int)(row >> 16);            // 0=q rows, 1=k rows (65536 each)
  long long idx = row & 65535;
  const float* src = qkvf + ((long long)which * 65536 + idx) * HD_;
  float x0 = src[lane], x1 = src[lane + 32];
  float ss = x0 * x0 + x1 * x1;
#pragma unroll
  for (int d = 16; d >= 1; d >>= 1) ss += __shfl_xor(ss, d, 32);
  float inv = 1.0f / fmaxf(sqrtf(ss), 1e-12f);
  bf16* dst = (which ? kb : qb) + idx * HD_;
  dst[lane]      = (bf16)(x0 * inv);
  dst[lane + 32] = (bf16)(x1 * inv);
}

// v: [B,H,S,64] fp32 -> vT [B,H,64,S] bf16 so PV B-fragments are contiguous.
__global__ void transpose_v_kernel(const float* __restrict__ qkvf,
                                   bf16* __restrict__ vT) {
  long long i = (long long)blockIdx.x * blockDim.x + threadIdx.x; // 4,194,304
  int hd = (int)(i & 63);
  long long t = i >> 6;
  int s = (int)(t & 2047);
  long long bh = t >> 11;
  vT[(bh * HD_ + hd) * S_LEN + s] = (bf16)qkvf[2ll * 4194304 + i];
}

// ---------------------------------------------------------------------------
// Flash-style causal attention. One wave (block of 32) per 16 query rows of
// one (b,h). Keys streamed in chunks of 32: QK^T = 4 wmma, PV = 4 wmma.
// Online softmax via half-wave shuffles (C-fragment rows live per 16-lane
// half). P transits LDS to be re-read in A-fragment layout.
// ---------------------------------------------------------------------------
__global__ void attn_kernel(const bf16* __restrict__ qb,
                            const bf16* __restrict__ kbuf,
                            const bf16* __restrict__ vT,
                            bf16* __restrict__ ob) {
  __shared__ bf16 pT[16 * 32];
  int lane = threadIdx.x;
  int tile = blockIdx.x;                   // 0..4095
  int qt = tile & 127;                     // query tile within sequence
  int bh = tile >> 7;                      // 0..31 = b*16+h
  int qs = qt << 4;
  int hf = lane >> 4, ln = lane & 15;

  const bf16* Q  = qb   + (long long)bh * S_LEN * HD_;
  const bf16* Kb = kbuf + (long long)bh * S_LEN * HD_;
  const bf16* Vt = vT   + (long long)bh * HD_ * S_LEN;

  v16bf aq0 = load_a_frag(Q, HD_, qs, 0, lane);
  v16bf aq1 = load_a_frag(Q, HD_, qs, 32, lane);

  v8f o[4] = {};
  float rm[8], rs[8];
#pragma unroll
  for (int r = 0; r < 8; ++r) { rm[r] = -1e30f; rs[r] = 0.0f; }

  int kend = qs + 16;                      // exclusive key bound (causal)
  for (int kc = 0; kc < kend; kc += 32) {
    __builtin_prefetch(Kb + (long long)(kc + 32) * HD_ + lane * 4, 0, 1);
    // --- sim tile [16 x 32] = Q @ K^T (K dim = 64, split into 2 wmma) ---
    v8f s0 = {}, s1 = {};
    s0 = wmma_bf16(aq0, load_b_frag(Kb, HD_, kc, 0, lane), s0);
    s0 = wmma_bf16(aq1, load_b_frag(Kb, HD_, kc, 32, lane), s0);
    s1 = wmma_bf16(aq0, load_b_frag(Kb, HD_, kc + 16, 0, lane), s1);
    s1 = wmma_bf16(aq1, load_b_frag(Kb, HD_, kc + 16, 32, lane), s1);

    int j0 = kc + ln, j1 = j0 + 16;
#pragma unroll
    for (int r = 0; r < 8; ++r) {
      int mabs = qs + r + (hf << 3);
      float v0 = s0[r] * SCALE_F;
      float v1 = s1[r] * SCALE_F;
      if (j0 > mabs) v0 = -1e30f;          // causal mask
      if (j1 > mabs) v1 = -1e30f;
      float mx = fmaxf(v0, v1);
#pragma unroll
      for (int d = 1; d < 16; d <<= 1) mx = fmaxf(mx, __shfl_xor(mx, d, 32));
      float nm = fmaxf(rm[r], mx);
      float alpha = __expf(rm[r] - nm);
      float p0 = __expf(v0 - nm);
      float p1 = __expf(v1 - nm);
      float ps = p0 + p1;
#pragma unroll
      for (int d = 1; d < 16; d <<= 1) ps += __shfl_xor(ps, d, 32);
      rs[r] = rs[r] * alpha + ps;
      rm[r] = nm;
      o[0][r] *= alpha; o[1][r] *= alpha; o[2][r] *= alpha; o[3][r] *= alpha;
      pT[(r + (hf << 3)) * 32 + ln]      = (bf16)p0;
      pT[(r + (hf << 3)) * 32 + ln + 16] = (bf16)p1;
    }
    __syncthreads();
    // re-read P in A-fragment layout (16 rows x 32 keys)
    ABf pa;
    const bf16* pp = pT + ln * 32 + (hf << 3);
    pa.h[0] = *reinterpret_cast<const v8bf*>(pp);
    pa.h[1] = *reinterpret_cast<const v8bf*>(pp + 16);
    __syncthreads();
    // --- O += P @ V  (V^T rows are head-dims, contiguous over keys) ---
#pragma unroll
    for (int t = 0; t < 4; ++t) {
      v16bf bv = load_b_frag(Vt, S_LEN, t << 4, kc, lane);
      o[t] = wmma_bf16(pa.v, bv, o[t]);
    }
  }

  // epilogue: normalize by row sums, emit bf16 in [B,S,D] layout for proj GEMM
  int h = bh & 15, b_ = bh >> 4;
#pragma unroll
  for (int r = 0; r < 8; ++r) {
    float inv = 1.0f / rs[r];
    int srow = qs + r + (hf << 3);
    bf16* dst = ob + ((long long)(b_ * S_LEN + srow)) * DIM + h * HD_;
#pragma unroll
    for (int t = 0; t < 4; ++t) dst[(t << 4) + ln] = (bf16)(o[t][r] * inv);
  }
}

// ---------------------------------------------------------------------------
extern "C" void kernel_launch(void* const* d_in, const int* in_sizes, int n_in,
                              void* d_out, int out_size, void* d_ws,
                              size_t ws_size, hipStream_t stream) {
  const float* x      = (const float*)d_in[0];
  const float* w_attn = (const float*)d_in[1];
  const float* b_attn = (const float*)d_in[2];
  const float* w_proj = (const float*)d_in[3];
  const float* b_proj = (const float*)d_in[4];
  float* out = (float*)d_out;

  char* ws = (char*)d_ws;
  size_t off = 0;
  auto alloc = [&](size_t bytes) -> void* {
    void* p = ws + off;
    off = (off + bytes + 255) & ~(size_t)255;
    return p;
  };
  bf16*  xb   = (bf16*)alloc(4194304ull * 2);   // x in bf16            (8 MB)
  bf16*  wabT = (bf16*)alloc(3145728ull * 2);   // w_attn^T bf16        (6 MB)
  bf16*  wpT  = (bf16*)alloc(1048576ull * 2);   // w_proj^T bf16        (2 MB)
  float* qkvf = (float*)alloc(12582912ull * 4); // [3][B][H][S][64]    (48 MB)
  bf16*  qb   = (bf16*)alloc(4194304ull * 2);   // normalized q bf16    (8 MB)
  bf16*  kb   = (bf16*)alloc(4194304ull * 2);   // normalized k bf16    (8 MB)
  bf16*  vT   = (bf16*)alloc(4194304ull * 2);   // v^T bf16 [B,H,64,S]  (8 MB)
  bf16*  ob   = (bf16*)alloc(4194304ull * 2);   // attn out bf16        (8 MB)

  convert_x_kernel<<<16384, 256, 0, stream>>>(x, xb);
  transpose_wattn_kernel<<<12288, 256, 0, stream>>>(w_attn, wabT);
  transpose_wproj_kernel<<<4096, 256, 0, stream>>>(w_proj, wpT);
  gemm_qkv_kernel<<<dim3(64, 48), 128, 0, stream>>>(xb, wabT, b_attn, qkvf);
  norm_qk_kernel<<<16384, 256, 0, stream>>>(qkvf, qb, kb);
  transpose_v_kernel<<<16384, 256, 0, stream>>>(qkvf, vT);
  attn_kernel<<<4096, 32, 0, stream>>>(qb, kb, vT, ob);
  gemm_proj_kernel<<<dim3(64, 16), 128, 0, stream>>>(ob, wpT, b_proj, out);
}
